// ContrastiveLoss_38792144618252
// MI455X (gfx1250) — matrix operand order repaired and checked by reading, not compiled, not run on hardware
//
#include <hip/hip_runtime.h>
#include <math.h>

#define N_PTS 8192
#define DIM   1024
#define INV_T (1.0f / 0.07f)

// B-staging geometry
#define KB      128                 // K-block halves staged per buffer
#define BPAD    8                   // pad halves per column (bank spread)
#define BSTRIDE (KB + BPAD)         // 136 halves per column in LDS
#define GCOLS   64                  // columns per group (4 WMMA tiles)
#define NSTAGE  (DIM / KB)          // 8 K-stages per column group

typedef _Float16 v8h  __attribute__((ext_vector_type(8)));
typedef _Float16 v16h __attribute__((ext_vector_type(16)));
typedef float    v8f  __attribute__((ext_vector_type(8)));

// Workspace layout (bytes):
//  fh (f16 hi)  : [0, 16MB)
//  fl (f16 lo)  : [16MB, 32MB)
//  denom (f32)  : N floats
//  possum (f32) : N floats
static constexpr size_t OFF_FH  = 0;
static constexpr size_t OFF_FL  = (size_t)N_PTS * DIM * 2;
static constexpr size_t OFF_DEN = OFF_FL * 2;
static constexpr size_t OFF_POS = OFF_DEN + (size_t)N_PTS * 4;

// ---------------------------------------------------------------------------
// Kernel 1: L2-normalize each row, split into f16 hi/lo pair.
// ---------------------------------------------------------------------------
__global__ void __launch_bounds__(256)
norm_split_kernel(const float* __restrict__ X,
                  _Float16* __restrict__ fh, _Float16* __restrict__ fl) {
  const int row = blockIdx.x;
  const int tid = threadIdx.x;
  const float* xr = X + (size_t)row * DIM;

  float v[4];
  float ss = 0.0f;
#pragma unroll
  for (int k = 0; k < 4; ++k) {
    v[k] = xr[tid + k * 256];
    ss += v[k] * v[k];
  }
  __shared__ float red[256];
  red[tid] = ss;
  __syncthreads();
  for (int s = 128; s > 0; s >>= 1) {
    if (tid < s) red[tid] += red[tid + s];
    __syncthreads();
  }
  const float scale = 1.0f / fmaxf(sqrtf(red[0]), 1e-12f);

#pragma unroll
  for (int k = 0; k < 4; ++k) {
    const float y = v[k] * scale;
    const _Float16 h = (_Float16)y;
    const _Float16 l = (_Float16)(y - (float)h);
    fh[(size_t)row * DIM + tid + k * 256] = h;
    fl[(size_t)row * DIM + tid + k * 256] = l;
  }
}

// ---------------------------------------------------------------------------
// Kernel 2: zero accumulators.
// ---------------------------------------------------------------------------
__global__ void zero_kernel(float* __restrict__ p, int n) {
  const int i = blockIdx.x * 256 + threadIdx.x;
  if (i < n) p[i] = 0.0f;
}

// ---------------------------------------------------------------------------
// Kernel 3: fused sim = (F F^T)/T with streaming epilogue.
// Block = 8 waves = 128 rows; all waves share one 64-column group whose B
// panel (hi+lo) is async-staged into LDS (double buffered) and consumed by
// every wave -> 8x less L2 traffic for B. Each wave: 16 rows x 4 tiles.
// The 12 WMMAs per K-step are issued round-robin over the 4 accumulator
// chains so same-accumulator WMMAs sit 4 apart (hides XDL latency).
// ---------------------------------------------------------------------------
__global__ void __launch_bounds__(256)
sim_kernel(const _Float16* __restrict__ fh, const _Float16* __restrict__ fl,
           const int* __restrict__ types,
           float* __restrict__ denom, float* __restrict__ possum) {
  // [buf][hi/lo][col * BSTRIDE + k]  : 2*2*64*136 halves = 68 KB
  __shared__ _Float16 sB[2][2][GCOLS * BSTRIDE];

  const int tid  = threadIdx.x;
  const int wv   = tid >> 5;
  const int lane = tid & 31;
  const int hgrp = lane >> 4;   // 0: lanes 0-15, 1: lanes 16-31
  const int l15  = lane & 15;

  const int rg        = blockIdx.x >> 2;           // 64 row groups of 128 rows
  const int ch        = blockIdx.x & 3;            // 4 column chunks of 2048
  const int row0      = (rg << 7) + (wv << 4);     // wave's 16-row strip
  const int chunkBase = ch << 11;

  // A operand: lane holds row (row0 + l15); upper lane-group K-shift = +16.
  const _Float16* aRowH = fh + (size_t)(row0 + l15) * DIM;
  const _Float16* aRowL = fl + (size_t)(row0 + l15) * DIM;
  const int aOff = hgrp << 4;

  int trow[8];
#pragma unroll
  for (int v = 0; v < 8; ++v) trow[v] = types[row0 + v + (hgrp << 3)];

  float dAcc[8], pAcc[8];
#pragma unroll
  for (int v = 0; v < 8; ++v) { dAcc[v] = 0.0f; pAcc[v] = 0.0f; }

  // Async stage: cooperatively copy B block (hi+lo) for cols [colBase,+64),
  // K [kb, kb+KB) into LDS buffer `buf`. 8 async b128 per thread per stage.
  auto issue_stage = [&](int buf, int colBase, int kb) {
#pragma unroll
    for (int i = 0; i < 4; ++i) {
      const int id = tid + (i << 8);   // 0..1023
      const int c  = id >> 4;          // column 0..63
      const int cb = id & 15;          // 16B chunk within K-block
      const _Float16* gH = fh + (size_t)(colBase + c) * DIM + kb + cb * 8;
      const _Float16* gL = fl + (size_t)(colBase + c) * DIM + kb + cb * 8;
      const unsigned lH = (unsigned)(size_t)&sB[buf][0][c * BSTRIDE + cb * 8];
      const unsigned lL = (unsigned)(size_t)&sB[buf][1][c * BSTRIDE + cb * 8];
      asm volatile("global_load_async_to_lds_b128 %0, %1, off"
                   :: "v"(lH), "v"(gH) : "memory");
      asm volatile("global_load_async_to_lds_b128 %0, %1, off"
                   :: "v"(lL), "v"(gL) : "memory");
    }
  };

  for (int cg = 0; cg < 32; ++cg) {
    const int colBase = chunkBase + (cg << 6);

    v8f c[4];
#pragma unroll
    for (int t = 0; t < 4; ++t) c[t] = (v8f){};

    issue_stage(0, colBase, 0);                       // prologue: buffer 0

    for (int st = 0; st < NSTAGE; ++st) {
      const int kb = st << 7;
      if (st + 1 < NSTAGE) {
        issue_stage((st + 1) & 1, colBase, kb + KB);  // prefetch next buffer
        asm volatile("s_wait_asynccnt 8" ::: "memory");   // stage st landed
      } else {
        asm volatile("s_wait_asynccnt 0" ::: "memory");
      }
      __syncthreads();

      const int bsel = st & 1;
      for (int kl = 0; kl < KB; kl += 32) {
        const int k = kb + kl;
        const v8h a0h = *(const v8h*)(aRowH + k + aOff);
        const v8h a1h = *(const v8h*)(aRowH + k + 8 + aOff);
        const v8h a0l = *(const v8h*)(aRowL + k + aOff);
        const v8h a1l = *(const v8h*)(aRowL + k + 8 + aOff);
        const v16h ah = __builtin_shufflevector(a0h, a1h,
            0,1,2,3,4,5,6,7,8,9,10,11,12,13,14,15);
        const v16h al = __builtin_shufflevector(a0l, a1l,
            0,1,2,3,4,5,6,7,8,9,10,11,12,13,14,15);

        // Load all 4 tiles' B operands up front, then round-robin WMMAs.
        v16h bh[4], bl[4];
#pragma unroll
        for (int t = 0; t < 4; ++t) {
          const int boff = (t * 16 + l15) * BSTRIDE + kl + (hgrp << 4);
          const _Float16* bhP = &sB[bsel][0][boff];
          const _Float16* blP = &sB[bsel][1][boff];
          const v8h bh0 = *(const v8h*)bhP;
          const v8h bh1 = *(const v8h*)(bhP + 8);
          const v8h bl0 = *(const v8h*)blP;
          const v8h bl1 = *(const v8h*)(blP + 8);
          bh[t] = __builtin_shufflevector(bh0, bh1,
              0,1,2,3,4,5,6,7,8,9,10,11,12,13,14,15);
          bl[t] = __builtin_shufflevector(bl0, bl1,
              0,1,2,3,4,5,6,7,8,9,10,11,12,13,14,15);
        }
#pragma unroll
        for (int t = 0; t < 4; ++t)
          c[t] = __builtin_amdgcn_wmma_f32_16x16x32_f16(false, ah, false, bh[t], (short)0, c[t], false, false);
#pragma unroll
        for (int t = 0; t < 4; ++t)
          c[t] = __builtin_amdgcn_wmma_f32_16x16x32_f16(false, ah, false, bl[t], (short)0, c[t], false, false);
#pragma unroll
        for (int t = 0; t < 4; ++t)
          c[t] = __builtin_amdgcn_wmma_f32_16x16x32_f16(false, al, false, bh[t], (short)0, c[t], false, false);
      }
      __syncthreads();
    }

    // Epilogue: mask diagonal, exp into denominator, masked sum for positives.
#pragma unroll
    for (int t = 0; t < 4; ++t) {
      const int j  = colBase + t * 16 + l15;
      const int tj = types[j];
#pragma unroll
      for (int v = 0; v < 8; ++v) {
        const int i = row0 + v + (hgrp << 3);
        const float s = c[t][v] * INV_T;
        if (i != j) {
          dAcc[v] += __expf(s);
          if (trow[v] == tj) pAcc[v] += s;
        }
      }
    }
  }

  // Row reduction across the 16 lanes of each lane-group, then one atomic
  // per row per column-chunk (4 adds per row total).
#pragma unroll
  for (int v = 0; v < 8; ++v) {
    float d = dAcc[v];
    float p = pAcc[v];
    for (int off = 8; off > 0; off >>= 1) {
      d += __shfl_xor(d, off, 32);
      p += __shfl_xor(p, off, 32);
    }
    if (l15 == 0) {
      const int i = row0 + v + (hgrp << 3);
      atomicAdd(&denom[i], d);
      atomicAdd(&possum[i], p);
    }
  }
}

// ---------------------------------------------------------------------------
// Kernel 4: type histogram -> pos_count, per-row loss, mean over valid rows.
// ---------------------------------------------------------------------------
__global__ void __launch_bounds__(256)
finalize_kernel(const int* __restrict__ types,
                const float* __restrict__ denom,
                const float* __restrict__ possum,
                float* __restrict__ out) {
  __shared__ int   hist[32];
  __shared__ float sTot[256];
  __shared__ float sCnt[256];
  const int tid = threadIdx.x;
  if (tid < 32) hist[tid] = 0;
  __syncthreads();
  for (int i = tid; i < N_PTS; i += 256) atomicAdd(&hist[types[i] & 31], 1);
  __syncthreads();

  float tot = 0.0f, cnt = 0.0f;
  for (int i = tid; i < N_PTS; i += 256) {
    const int pc = hist[types[i] & 31] - 1;
    if (pc > 0) {
      const float pm = possum[i] / (float)pc;
      const float li = -logf(expf(pm) / denom[i] + 1e-10f);
      tot += li;
      cnt += 1.0f;
    }
  }
  sTot[tid] = tot;
  sCnt[tid] = cnt;
  __syncthreads();
  for (int s = 128; s > 0; s >>= 1) {
    if (tid < s) { sTot[tid] += sTot[tid + s]; sCnt[tid] += sCnt[tid + s]; }
    __syncthreads();
  }
  if (tid == 0) out[0] = (sCnt[0] > 0.0f) ? (sTot[0] / sCnt[0]) : 0.0f;
}

// ---------------------------------------------------------------------------
extern "C" void kernel_launch(void* const* d_in, const int* in_sizes, int n_in,
                              void* d_out, int out_size, void* d_ws, size_t ws_size,
                              hipStream_t stream) {
  const float* X     = (const float*)d_in[0];
  const int*   types = (const int*)d_in[1];
  char* ws = (char*)d_ws;
  _Float16* fh     = (_Float16*)(ws + OFF_FH);
  _Float16* fl     = (_Float16*)(ws + OFF_FL);
  float*    denom  = (float*)(ws + OFF_DEN);
  float*    possum = (float*)(ws + OFF_POS);

  norm_split_kernel<<<N_PTS, 256, 0, stream>>>(X, fh, fl);
  zero_kernel<<<(2 * N_PTS + 255) / 256, 256, 0, stream>>>(denom, 2 * N_PTS);
  sim_kernel<<<256, 256, 0, stream>>>(fh, fl, types, denom, possum);
  finalize_kernel<<<1, 256, 0, stream>>>(types, denom, possum, (float*)d_out);
}
